// TransformerBlock_38912403702358
// MI455X (gfx1250) — matrix operand order repaired and checked by reading, not compiled, not run on hardware
//
#include <hip/hip_runtime.h>
#include <hip/hip_bf16.h>

// ---------------- problem constants ----------------
#define Bn    2
#define Sn    2048
#define Dn    1024
#define NHn   16
#define NKVn  4
#define HDn   64
#define NREPn (NHn / NKVn)
#define En    8
#define TOPKn 2
#define HIDn  1365            // 4*D//3
#define HIDP  1376            // HID padded to multiple of 8 (16B rows for bf16)
#define Tn    (Bn * Sn)       // 4096 tokens
#define CAPACITYn 1280        // int(T*TOPK/E*1.25)

typedef __attribute__((ext_vector_type(16))) __bf16 v16bf;
typedef __attribute__((ext_vector_type(8)))  __bf16 v8bf;
typedef __attribute__((ext_vector_type(8)))  float  v8f;

// ---------------- helpers ----------------
__device__ __forceinline__ __bf16 f2bf(float f) {
  union { float f; unsigned u; } v; v.f = f;
  unsigned r = v.u + 0x7FFFu + ((v.u >> 16) & 1u);   // round-nearest-even
  union { unsigned short s; __bf16 b; } o; o.s = (unsigned short)(r >> 16);
  return o.b;
}
__device__ __forceinline__ __bf16 bf_zero() {
  union { unsigned short s; __bf16 b; } o; o.s = 0; return o.b;
}

__device__ __forceinline__ v16bf ld_frag2(const __bf16* p0, const __bf16* p1) {
  const v8bf lo = *(const v8bf*)p0;
  const v8bf hi = *(const v8bf*)p1;
  v16bf r;
#pragma unroll
  for (int i = 0; i < 8; i++) { r[i] = lo[i]; r[i + 8] = hi[i]; }
  return r;
}
__device__ __forceinline__ v16bf ld_frag16(const __bf16* p) { return ld_frag2(p, p + 8); }

__device__ __forceinline__ v8f wmma_bf16(v16bf a, v16bf b, v8f c) {
  return __builtin_amdgcn_wmma_f32_16x16x32_bf16(false, a, false, b, (short)0, c, false, false);
}

// ---------------- generic bf16 WMMA GEMM ----------------
// C[M,N] = A[M,K] * B[K,N], with B supplied PRE-TRANSPOSED as Bt[N][ldb] (ldb >= K).
// A has leading dimension lda (>= K), C has leading dimension ldc (>= N).
// Double-buffered LDS pipeline: global loads of tile t+1 overlap WMMA on tile t.
#define BM 128
#define BN 128
#define BK 32

// Load a 128x32 bf16 tile (rows r0.., k-cols k0..) into 2 v8bf registers/thread.
__device__ __forceinline__ void tile_to_regs(const __bf16* __restrict__ P, int ld,
                                             int r0, int k0, int Rmax, int Kmax,
                                             int tid, v8bf out[2]) {
#pragma unroll
  for (int g2 = 0; g2 < 2; g2++) {
    int g = tid + g2 * 256;            // 0..511
    int row = g >> 2, cg = g & 3;
    int gr = r0 + row, gk = k0 + cg * 8;
    v8bf v;
    if (gr < Rmax && gk + 8 <= Kmax) {
      v = *(const v8bf*)&P[(size_t)gr * ld + gk];
    } else {
#pragma unroll
      for (int j = 0; j < 8; j++)
        v[j] = (gr < Rmax && gk + j < Kmax) ? P[(size_t)gr * ld + gk + j] : bf_zero();
    }
    out[g2] = v;
  }
}

__device__ __forceinline__ void regs_to_lds(__bf16 (*S)[BK + 8], int tid, const v8bf in[2]) {
#pragma unroll
  for (int g2 = 0; g2 < 2; g2++) {
    int g = tid + g2 * 256;
    int row = g >> 2, cg = g & 3;
    *(v8bf*)&S[row][cg * 8] = in[g2];
  }
}

__global__ __launch_bounds__(256) void k_gemm_bf16(const __bf16* __restrict__ A, int lda,
                                                   const __bf16* __restrict__ Bt, int ldb,
                                                   float* __restrict__ C, int ldc,
                                                   int M, int N, int K) {
  __builtin_amdgcn_s_wait_tensorcnt(0);
  __shared__ __attribute__((aligned(16))) __bf16 As[2][BM][BK + 8];
  __shared__ __attribute__((aligned(16))) __bf16 Bs[2][BN][BK + 8];   // [n][k]
  const int tid = threadIdx.x;
  const int w = tid >> 5, lane = tid & 31;
  const int hl = lane >> 4, r = lane & 15;
  const int wm = w >> 1, wn = w & 1;                 // 4 x 2 wave grid -> 32x64 per wave
  const int m0 = blockIdx.x * BM, n0 = blockIdx.y * BN;
  v8f acc[2][4] = {};
  v8bf ra[2], rb[2];

  // prologue: stage tile 0
  tile_to_regs(A, lda, m0, 0, M, K, tid, ra);
  tile_to_regs(Bt, ldb, n0, 0, N, K, tid, rb);
  regs_to_lds(As[0], tid, ra);
  regs_to_lds(Bs[0], tid, rb);
  __syncthreads();

  const int nk = (K + BK - 1) / BK;
  for (int t = 0; t < nk; t++) {
    const int cur = t & 1, nxt = cur ^ 1;
    const bool hasNext = (t + 1 < nk);
    if (hasNext) {                                  // issue global loads for t+1
      tile_to_regs(A, lda, m0, (t + 1) * BK, M, K, tid, ra);
      tile_to_regs(Bt, ldb, n0, (t + 1) * BK, N, K, tid, rb);
      if (t + 2 < nk) {                             // global_prefetch_b8 two tiles ahead
        int pr = m0 + (tid >> 1);
        if (pr < M) __builtin_prefetch(&A[(size_t)pr * lda + (t + 2) * BK], 0, 1);
        int pn = n0 + (tid >> 1);
        if (pn < N) __builtin_prefetch(&Bt[(size_t)pn * ldb + (t + 2) * BK], 0, 1);
      }
    }
    // WMMA on current LDS buffers
    v16bf bfr[4];
#pragma unroll
    for (int ni = 0; ni < 4; ni++)
      bfr[ni] = ld_frag2(&Bs[cur][wn * 64 + ni * 16 + r][8 * hl], &Bs[cur][wn * 64 + ni * 16 + r][16 + 8 * hl]);
#pragma unroll
    for (int mi = 0; mi < 2; mi++) {
      const __bf16* pa = &As[cur][wm * 32 + mi * 16 + r][8 * hl];
      v16bf a = ld_frag2(pa, pa + 16);
#pragma unroll
      for (int ni = 0; ni < 4; ni++)
        acc[mi][ni] = wmma_bf16(a, bfr[ni], acc[mi][ni]);
    }
    if (hasNext) {
      regs_to_lds(As[nxt], tid, ra);
      regs_to_lds(Bs[nxt], tid, rb);
    }
    __syncthreads();
  }
#pragma unroll
  for (int mi = 0; mi < 2; mi++)
#pragma unroll
    for (int ni = 0; ni < 4; ni++)
#pragma unroll
      for (int i = 0; i < 8; i++) {
        int row = m0 + wm * 32 + mi * 16 + i + 8 * hl;
        int col = n0 + wn * 64 + ni * 16 + r;
        if (row < M && col < N) C[(size_t)row * ldc + col] = acc[mi][ni][i];
      }
}

// ---------------- transpose + f32->bf16 convert: W[K][N] -> Wt[N][ldb] ----------------
__global__ __launch_bounds__(256) void k_cvt_T(const float* __restrict__ W,
                                               __bf16* __restrict__ Wt,
                                               int K, int N, int ldb) {
  __shared__ float tile[32][33];
  int tx = threadIdx.x & 31, ty = threadIdx.x >> 5;
  int k0 = blockIdx.x * 32, n0 = blockIdx.y * 32;
  for (int i = ty; i < 32; i += 8) {
    int gk = k0 + i, gn = n0 + tx;
    tile[i][tx] = (gk < K && gn < N) ? W[(size_t)gk * N + gn] : 0.f;
  }
  __syncthreads();
  for (int i = ty; i < 32; i += 8) {
    int gn = n0 + i, gk = k0 + tx;
    if (gn < N && gk < ldb) Wt[(size_t)gn * ldb + gk] = f2bf(tile[tx][i]);
  }
}

// ---------------- flash attention (bf16 WMMA, online softmax) ----------------
__global__ __launch_bounds__(128) void k_attn(const __bf16* __restrict__ Q,   // [B][NH][S][HD]
                                              const __bf16* __restrict__ Kc,  // [B][NKV][S][HD]
                                              const __bf16* __restrict__ Vc,  // [B][NKV][S][HD]
                                              __bf16* __restrict__ ao) {      // [B][S][NH*HD]
  const int bh = blockIdx.x;
  const int b = bh / NHn, h = bh % NHn, kvh = h / NREPn;
  const int w = threadIdx.x >> 5, lane = threadIdx.x & 31;
  const int hl = lane >> 4, r = lane & 15;
  const int qbase = blockIdx.y * 64 + w * 16;

  __shared__ __attribute__((aligned(16))) __bf16 Vt[64][40];        // [d][k] chunk of 32 kv
  __shared__ __attribute__((aligned(16))) __bf16 Pl[4][16][40];     // per-wave P tile

  const __bf16* qp = Q + ((size_t)b * NHn + h) * Sn * HDn;
  const __bf16* kp = Kc + ((size_t)b * NKVn + kvh) * Sn * HDn;
  const __bf16* vp = Vc + ((size_t)b * NKVn + kvh) * Sn * HDn;

  v16bf aq[2];
#pragma unroll
  for (int c = 0; c < 2; c++) {
    const __bf16* p = qp + (size_t)(qbase + r) * HDn + c * 32 + 8 * hl;
    aq[c] = ld_frag2(p, p + 16);
  }
  float m[8], l[8];
  v8f acc[4] = {};
#pragma unroll
  for (int i = 0; i < 8; i++) { m[i] = -1e30f; l[i] = 0.f; }

  const int kend = blockIdx.y * 64 + 64;
  for (int kc = 0; kc < kend; kc += 32) {
    // cooperative V chunk staging, vectorized 16B reads, transpose into LDS
    for (int g = threadIdx.x; g < 32 * 8; g += 128) {
      int kk = g >> 3, dg = g & 7;
      v8bf vv = *(const v8bf*)&vp[(size_t)(kc + kk) * HDn + dg * 8];
#pragma unroll
      for (int j = 0; j < 8; j++) Vt[dg * 8 + j][kk] = vv[j];
    }
    __syncthreads();
    if (kc <= qbase + 15) {              // wave-uniform: skip fully-masked chunks
      v8f s[2] = {};
#pragma unroll
      for (int t = 0; t < 2; t++) {
        int kcol = kc + t * 16 + r;
#pragma unroll
        for (int c = 0; c < 2; c++) {
          v16bf bk = ld_frag2(kp + (size_t)kcol * HDn + c * 32 + 8 * hl, kp + (size_t)kcol * HDn + c * 32 + 16 + 8 * hl);
          s[t] = wmma_bf16(aq[c], bk, s[t]);
        }
      }
      float cm[8];
#pragma unroll
      for (int i = 0; i < 8; i++) {
        int qrow = qbase + i + 8 * hl;
        float v0 = s[0][i] * 0.125f + ((kc + r) > qrow ? -1e9f : 0.f);
        float v1 = s[1][i] * 0.125f + ((kc + 16 + r) > qrow ? -1e9f : 0.f);
        s[0][i] = v0; s[1][i] = v1;
        cm[i] = fmaxf(v0, v1);
      }
#pragma unroll
      for (int i = 0; i < 8; i++)
#pragma unroll
        for (int off = 1; off < 16; off <<= 1)
          cm[i] = fmaxf(cm[i], __shfl_xor(cm[i], off, 32));
#pragma unroll
      for (int i = 0; i < 8; i++) {
        float mn = fmaxf(m[i], cm[i]);
        float sc = __expf(m[i] - mn);
        m[i] = mn;
        float p0 = __expf(s[0][i] - mn), p1 = __expf(s[1][i] - mn);
        Pl[w][i + 8 * hl][r] = f2bf(p0);
        Pl[w][i + 8 * hl][16 + r] = f2bf(p1);
        float rs = p0 + p1;
#pragma unroll
        for (int off = 1; off < 16; off <<= 1) rs += __shfl_xor(rs, off, 32);
        l[i] = l[i] * sc + rs;
#pragma unroll
        for (int nt = 0; nt < 4; nt++) acc[nt][i] *= sc;
      }
      v16bf pf = ld_frag2(&Pl[w][r][8 * hl], &Pl[w][r][16 + 8 * hl]);
#pragma unroll
      for (int nt = 0; nt < 4; nt++) {
        v16bf vf = ld_frag2(&Vt[nt * 16 + r][8 * hl], &Vt[nt * 16 + r][16 + 8 * hl]);
        acc[nt] = wmma_bf16(pf, vf, acc[nt]);
      }
    }
    __syncthreads();
  }
#pragma unroll
  for (int i = 0; i < 8; i++) {
    float inv = 1.f / l[i];
    int qrow = qbase + i + 8 * hl;
    size_t base = ((size_t)b * Sn + qrow) * (size_t)Dn + h * HDn;
#pragma unroll
    for (int nt = 0; nt < 4; nt++)
      ao[base + nt * 16 + r] = f2bf(acc[nt][i] * inv);
  }
}

// ---------------- elementwise / small kernels ----------------
__global__ void k_zerof(float* __restrict__ p, size_t n) {
  size_t i = (size_t)blockIdx.x * blockDim.x + threadIdx.x;
  size_t st = (size_t)gridDim.x * blockDim.x;
  for (; i < n; i += st) p[i] = 0.f;
}

__global__ __launch_bounds__(256) void k_rmsnorm(const float* __restrict__ x,
                                                 const float* __restrict__ w,
                                                 __bf16* __restrict__ obf,
                                                 float* __restrict__ of) {
  int row = blockIdx.x;
  const float* xr = x + (size_t)row * Dn;
  __shared__ float red[256];
  float ss = 0.f;
  for (int d = threadIdx.x; d < Dn; d += 256) { float v = xr[d]; ss += v * v; }
  red[threadIdx.x] = ss; __syncthreads();
  for (int off = 128; off > 0; off >>= 1) {
    if (threadIdx.x < off) red[threadIdx.x] += red[threadIdx.x + off];
    __syncthreads();
  }
  float inv = rsqrtf(red[0] / (float)Dn + 1e-6f);
  for (int d = threadIdx.x; d < Dn; d += 256) {
    float y = xr[d] * inv * w[d];
    obf[(size_t)row * Dn + d] = f2bf(y);
    if (of) of[(size_t)row * Dn + d] = y;
  }
}

__global__ void k_rope_q(const float* __restrict__ q, const float* __restrict__ cs,
                         const float* __restrict__ sn, __bf16* __restrict__ qb) {
  size_t i = (size_t)blockIdx.x * blockDim.x + threadIdx.x;
  size_t n = (size_t)Tn * NHn * 32;
  size_t st = (size_t)gridDim.x * blockDim.x;
  for (; i < n; i += st) {
    int j = (int)(i & 31);
    size_t t2 = i >> 5;
    int h = (int)(t2 % NHn);
    size_t t = t2 / NHn;
    int s = (int)(t % Sn), b = (int)(t / Sn);
    float x1 = q[t * Dn + h * HDn + j];
    float x2 = q[t * Dn + h * HDn + 32 + j];
    float ch = cs[s * HDn + j], sh = sn[s * HDn + j];
    size_t base = (((size_t)b * NHn + h) * Sn + s) * HDn;
    qb[base + j]      = f2bf(x1 * ch - x2 * sh);
    qb[base + 32 + j] = f2bf(x1 * sh + x2 * ch);
  }
}

__global__ void k_rope_k(const float* __restrict__ kin, const float* __restrict__ cs,
                         const float* __restrict__ sn, float* __restrict__ newk,
                         __bf16* __restrict__ kb) {
  size_t i = (size_t)blockIdx.x * blockDim.x + threadIdx.x;
  size_t n = (size_t)Tn * NKVn * 32;
  size_t st = (size_t)gridDim.x * blockDim.x;
  for (; i < n; i += st) {
    int j = (int)(i & 31);
    size_t t2 = i >> 5;
    int kv = (int)(t2 % NKVn);
    size_t t = t2 / NKVn;
    int s = (int)(t % Sn), b = (int)(t / Sn);
    float x1 = kin[t * (NKVn * HDn) + kv * HDn + j];
    float x2 = kin[t * (NKVn * HDn) + kv * HDn + 32 + j];
    float ch = cs[s * HDn + j], sh = sn[s * HDn + j];
    float o1 = x1 * ch - x2 * sh, o2 = x1 * sh + x2 * ch;
    newk[t * (NKVn * HDn) + kv * HDn + j] = o1;
    newk[t * (NKVn * HDn) + kv * HDn + 32 + j] = o2;
    size_t base = (((size_t)b * NKVn + kv) * Sn + s) * HDn;
    kb[base + j] = f2bf(o1);
    kb[base + 32 + j] = f2bf(o2);
  }
}

__global__ void k_trans_v(const float* __restrict__ v, __bf16* __restrict__ vb) {
  size_t i = (size_t)blockIdx.x * blockDim.x + threadIdx.x;
  size_t n = (size_t)Tn * NKVn * 8;            // groups of 8 contiguous d
  size_t st = (size_t)gridDim.x * blockDim.x;
  for (; i < n; i += st) {
    int dg = (int)(i & 7);
    int kv = (int)((i >> 3) & 3);
    size_t t = i >> 5;
    int s = (int)(t % Sn), b = (int)(t / Sn);
    const float* src = v + t * (NKVn * HDn) + kv * HDn + dg * 8;
    v8bf o;
#pragma unroll
    for (int j = 0; j < 8; j++) o[j] = f2bf(src[j]);
    *(v8bf*)&vb[(((size_t)b * NKVn + kv) * Sn + s) * HDn + dg * 8] = o;
  }
}

__global__ void k_add(const float* __restrict__ a, const float* __restrict__ b,
                      float* __restrict__ o, size_t n) {
  size_t i = (size_t)blockIdx.x * blockDim.x + threadIdx.x;
  size_t st = (size_t)gridDim.x * blockDim.x;
  for (; i < n; i += st) o[i] = a[i] + b[i];
}

__global__ void k_silu_mul(const float* __restrict__ h1, const float* __restrict__ h3,
                           __bf16* __restrict__ u, size_t n) {
  size_t i = (size_t)blockIdx.x * blockDim.x + threadIdx.x;
  size_t st = (size_t)gridDim.x * blockDim.x;
  for (; i < n; i += st) {
    float a = h1[i];
    float s = a / (1.f + __expf(-a));
    u[i] = f2bf(s * h3[i]);
  }
}

__global__ void k_moe_accum(float* __restrict__ moe, const float* __restrict__ eo,
                            const float* __restrict__ gate, int e) {
  size_t i = (size_t)blockIdx.x * blockDim.x + threadIdx.x;
  size_t n = (size_t)Tn * Dn;
  size_t st = (size_t)gridDim.x * blockDim.x;
  for (; i < n; i += st) {
    size_t t = i / Dn;
    float g = gate[t * En + e];
    if (g != 0.f) moe[i] += g * eo[i];
  }
}

__global__ void k_final(const float* __restrict__ h, const float* __restrict__ moe,
                        const float* __restrict__ so, float* __restrict__ out, size_t n) {
  size_t i = (size_t)blockIdx.x * blockDim.x + threadIdx.x;
  size_t st = (size_t)gridDim.x * blockDim.x;
  for (; i < n; i += st) out[i] = h[i] + moe[i] + so[i];
}

// ---------------- router / MoE gating ----------------
__global__ void k_zero_stats(float* psum, int* counts) {
  if (threadIdx.x < En) { psum[threadIdx.x] = 0.f; counts[threadIdx.x] = 0; }
}

__global__ __launch_bounds__(128) void k_router(const float* __restrict__ hn,
                                                const float* __restrict__ rw,
                                                float* __restrict__ gate,
                                                float* __restrict__ psum,
                                                int* __restrict__ counts) {
  int t = blockIdx.x;
  float p[En];
#pragma unroll
  for (int e = 0; e < En; e++) p[e] = 0.f;
  for (int d = threadIdx.x; d < Dn; d += 128) {
    float xv = hn[(size_t)t * Dn + d];
#pragma unroll
    for (int e = 0; e < En; e++) p[e] += xv * rw[(size_t)d * En + e];
  }
  __shared__ float red[En * 128];
#pragma unroll
  for (int e = 0; e < En; e++) red[e * 128 + threadIdx.x] = p[e];
  __syncthreads();
  for (int off = 64; off > 0; off >>= 1) {
    if (threadIdx.x < off) {
#pragma unroll
      for (int e = 0; e < En; e++)
        red[e * 128 + threadIdx.x] += red[e * 128 + threadIdx.x + off];
    }
    __syncthreads();
  }
  if (threadIdx.x == 0) {
    float pr[En];
    float mx = -1e30f;
#pragma unroll
    for (int e = 0; e < En; e++) mx = fmaxf(mx, red[e * 128]);
    float sum = 0.f;
#pragma unroll
    for (int e = 0; e < En; e++) { pr[e] = __expf(red[e * 128] - mx); sum += pr[e]; }
#pragma unroll
    for (int e = 0; e < En; e++) pr[e] /= sum;
    int i1 = 0;
#pragma unroll
    for (int e = 1; e < En; e++) if (pr[e] > pr[i1]) i1 = e;
    int i2 = (i1 == 0) ? 1 : 0;
#pragma unroll
    for (int e = 0; e < En; e++) if (e != i1 && pr[e] > pr[i2]) i2 = e;
    float w1 = pr[i1], w2 = pr[i2], wnm = w1 + w2;
#pragma unroll
    for (int e = 0; e < En; e++) gate[(size_t)t * En + e] = 0.f;
    gate[(size_t)t * En + i1] = w1 / wnm;
    gate[(size_t)t * En + i2] = w2 / wnm;
    atomicAdd(&counts[i1], 1);
    atomicAdd(&counts[i2], 1);
#pragma unroll
    for (int e = 0; e < En; e++) atomicAdd(&psum[e], pr[e]);
  }
}

__global__ __launch_bounds__(256) void k_capacity(float* __restrict__ gate, int capacity) {
  int e = blockIdx.x;
  __shared__ int sb[256];
  __shared__ int srun;
  if (threadIdx.x == 0) srun = 0;
  __syncthreads();
  for (int base = 0; base < Tn; base += 256) {
    int t = base + threadIdx.x;
    int mem = (gate[(size_t)t * En + e] > 0.f) ? 1 : 0;
    sb[threadIdx.x] = mem;
    __syncthreads();
    for (int off = 1; off < 256; off <<= 1) {
      int v = (threadIdx.x >= off) ? sb[threadIdx.x - off] : 0;
      __syncthreads();
      sb[threadIdx.x] += v;
      __syncthreads();
    }
    int cum = srun + sb[threadIdx.x];
    if (mem && cum > capacity) gate[(size_t)t * En + e] = 0.f;
    __syncthreads();
    if (threadIdx.x == 0) srun += sb[255];
    __syncthreads();
  }
}

__global__ void k_stats(const float* psum, const int* counts, float* aux, float* util) {
  if (threadIdx.x == 0 && blockIdx.x == 0) {
    float a = 0.f; int nz = 0;
#pragma unroll
    for (int e = 0; e < En; e++) {
      a += (psum[e] / (float)Tn) * ((float)counts[e] / (float)Tn);
      if (counts[e] > 0) nz++;
    }
    *aux = (float)En * a;
    *util = 100.f * (float)nz / (float)En;
  }
}

// ---------------- host orchestration ----------------
extern "C" void kernel_launch(void* const* d_in, const int* in_sizes, int n_in,
                              void* d_out, int out_size, void* d_ws, size_t ws_size,
                              hipStream_t stream) {
  const float* x   = (const float*)d_in[0];
  const float* cs  = (const float*)d_in[1];
  const float* sn  = (const float*)d_in[2];
  const float* anw = (const float*)d_in[4];
  const float* mnw = (const float*)d_in[5];
  const float* wq  = (const float*)d_in[6];
  const float* wk  = (const float*)d_in[7];
  const float* wv  = (const float*)d_in[8];
  const float* wo  = (const float*)d_in[9];
  const float* rw  = (const float*)d_in[10];
  const float* ew1 = (const float*)d_in[11];
  const float* ew2 = (const float*)d_in[12];
  const float* ew3 = (const float*)d_in[13];
  const float* sw1 = (const float*)d_in[14];
  const float* sw2 = (const float*)d_in[15];
  const float* sw3 = (const float*)d_in[16];

  float* dout = (float*)d_out;
  const size_t OUTSZ = (size_t)Tn * Dn;
  float* aux  = dout + OUTSZ;
  float* util = dout + OUTSZ + 1;
  float* newk = dout + OUTSZ + 2;
  float* newv = dout + OUTSZ + 2 + (size_t)Tn * NKVn * HDn;

  char* ws = (char*)d_ws;
  size_t off = 0;
  auto alloc = [&](size_t bytes) -> void* {
    off = (off + 255) & ~(size_t)255;
    void* p = ws + off;
    off += bytes;
    return p;
  };

  // transposed bf16 weights: [N][ld]
  __bf16* wqt  = (__bf16*)alloc((size_t)Dn * Dn * 2);                 // [1024][1024]
  __bf16* wkt  = (__bf16*)alloc((size_t)NKVn * HDn * Dn * 2);         // [256][1024]
  __bf16* wvt  = (__bf16*)alloc((size_t)NKVn * HDn * Dn * 2);
  __bf16* wot  = (__bf16*)alloc((size_t)Dn * Dn * 2);
  __bf16* ew1t = (__bf16*)alloc((size_t)En * HIDn * Dn * 2);          // per-e [HID][1024]
  __bf16* ew2t = (__bf16*)alloc((size_t)En * Dn * HIDP * 2);          // per-e [1024][HIDP]
  __bf16* ew3t = (__bf16*)alloc((size_t)En * HIDn * Dn * 2);
  __bf16* sw1t = (__bf16*)alloc((size_t)HIDn * Dn * 2);
  __bf16* sw2t = (__bf16*)alloc((size_t)Dn * HIDP * 2);
  __bf16* sw3t = (__bf16*)alloc((size_t)HIDn * Dn * 2);
  // activations
  __bf16* xnbf  = (__bf16*)alloc((size_t)Tn * Dn * 2);
  float*  qf    = (float*)alloc((size_t)Tn * Dn * 4);
  float*  kf    = (float*)alloc((size_t)Tn * NKVn * HDn * 4);
  __bf16* qbf   = (__bf16*)alloc((size_t)Tn * Dn * 2);
  __bf16* kbf   = (__bf16*)alloc((size_t)Tn * NKVn * HDn * 2);
  __bf16* vbf   = (__bf16*)alloc((size_t)Tn * NKVn * HDn * 2);
  __bf16* aobf  = (__bf16*)alloc((size_t)Tn * Dn * 2);
  float*  aproj = (float*)alloc((size_t)Tn * Dn * 4);
  float*  hbuf  = (float*)alloc((size_t)Tn * Dn * 4);
  float*  hnf   = (float*)alloc((size_t)Tn * Dn * 4);
  __bf16* hnbf  = (__bf16*)alloc((size_t)Tn * Dn * 2);
  float*  gate  = (float*)alloc((size_t)Tn * En * 4);
  float*  psum  = (float*)alloc(En * 4);
  int*    cnts  = (int*)alloc(En * 4);
  float*  h1    = (float*)alloc((size_t)Tn * HIDP * 4);
  float*  h3    = (float*)alloc((size_t)Tn * HIDP * 4);
  __bf16* ubf   = (__bf16*)alloc((size_t)Tn * HIDP * 2);
  float*  eo    = (float*)alloc((size_t)Tn * Dn * 4);
  float*  moe   = (float*)alloc((size_t)Tn * Dn * 4);
  (void)ws_size; (void)in_sizes; (void)n_in; (void)out_size;

  auto cvtT = [&](const float* W, __bf16* Wt, int K, int N, int ldb) {
    dim3 g((K + 31) / 32, (N + 31) / 32);
    k_cvt_T<<<g, 256, 0, stream>>>(W, Wt, K, N, ldb);
  };
  auto gemm = [&](const __bf16* A, int lda, const __bf16* Bt, int ldb,
                  float* C, int ldc, int M, int N, int K) {
    dim3 g((M + BM - 1) / BM, (N + BN - 1) / BN);
    k_gemm_bf16<<<g, 256, 0, stream>>>(A, lda, Bt, ldb, C, ldc, M, N, K);
  };

  // 1) weight transpose+convert
  cvtT(wq, wqt, Dn, Dn, Dn);
  cvtT(wk, wkt, Dn, NKVn * HDn, Dn);
  cvtT(wv, wvt, Dn, NKVn * HDn, Dn);
  cvtT(wo, wot, Dn, Dn, Dn);
  for (int e = 0; e < En; e++) {
    cvtT(ew1 + (size_t)e * Dn * HIDn, ew1t + (size_t)e * HIDn * Dn, Dn, HIDn, Dn);
    cvtT(ew3 + (size_t)e * Dn * HIDn, ew3t + (size_t)e * HIDn * Dn, Dn, HIDn, Dn);
    cvtT(ew2 + (size_t)e * HIDn * Dn, ew2t + (size_t)e * Dn * HIDP, HIDn, Dn, HIDP);
  }
  cvtT(sw1, sw1t, Dn, HIDn, Dn);
  cvtT(sw3, sw3t, Dn, HIDn, Dn);
  cvtT(sw2, sw2t, HIDn, Dn, HIDP);

  // 2) attention-input rmsnorm
  k_rmsnorm<<<Tn, 256, 0, stream>>>(x, anw, xnbf, (float*)nullptr);

  // 3) QKV projections (WMMA)
  gemm(xnbf, Dn, wqt, Dn, qf, NHn * HDn, Tn, NHn * HDn, Dn);
  gemm(xnbf, Dn, wkt, Dn, kf, NKVn * HDn, Tn, NKVn * HDn, Dn);
  gemm(xnbf, Dn, wvt, Dn, newv, NKVn * HDn, Tn, NKVn * HDn, Dn);   // v -> d_out new_v

  // 4) RoPE + head-major bf16 repacks (+ new_k output)
  k_rope_q<<<2048, 256, 0, stream>>>(qf, cs, sn, qbf);
  k_rope_k<<<1024, 256, 0, stream>>>(kf, cs, sn, newk, kbf);
  k_trans_v<<<512, 256, 0, stream>>>(newv, vbf);

  // 5) flash attention (WMMA)
  {
    dim3 g(Bn * NHn, Sn / 64);
    k_attn<<<g, 128, 0, stream>>>(qbf, kbf, vbf, aobf);
  }

  // 6) output projection + residual
  gemm(aobf, Dn, wot, Dn, aproj, Dn, Tn, Dn, Dn);
  k_add<<<2048, 256, 0, stream>>>(x, aproj, hbuf, (size_t)Tn * Dn);

  // 7) MoE-input rmsnorm (f32 for router, bf16 for GEMMs)
  k_rmsnorm<<<Tn, 256, 0, stream>>>(hbuf, mnw, hnbf, hnf);

  // 8) router, gating, capacity, stats
  k_zero_stats<<<1, 64, 0, stream>>>(psum, cnts);
  k_zerof<<<2048, 256, 0, stream>>>(moe, (size_t)Tn * Dn);
  k_router<<<Tn, 128, 0, stream>>>(hnf, rw, gate, psum, cnts);
  k_capacity<<<En, 256, 0, stream>>>(gate, CAPACITYn);
  k_stats<<<1, 1, 0, stream>>>(psum, cnts, aux, util);

  // 9) experts (dense compute per reference, gated accumulate)
  for (int e = 0; e < En; e++) {
    gemm(hnbf, Dn, ew1t + (size_t)e * HIDn * Dn, Dn, h1, HIDP, Tn, HIDn, Dn);
    gemm(hnbf, Dn, ew3t + (size_t)e * HIDn * Dn, Dn, h3, HIDP, Tn, HIDn, Dn);
    k_silu_mul<<<2048, 256, 0, stream>>>(h1, h3, ubf, (size_t)Tn * HIDP);
    gemm(ubf, HIDP, ew2t + (size_t)e * Dn * HIDP, HIDP, eo, Dn, Tn, Dn, HIDn);
    k_moe_accum<<<2048, 256, 0, stream>>>(moe, eo, gate, e);
  }

  // 10) shared expert (reuses h1/h3/ubf/eo)
  gemm(hnbf, Dn, sw1t, Dn, h1, HIDP, Tn, HIDn, Dn);
  gemm(hnbf, Dn, sw3t, Dn, h3, HIDP, Tn, HIDn, Dn);
  k_silu_mul<<<2048, 256, 0, stream>>>(h1, h3, ubf, (size_t)Tn * HIDP);
  gemm(ubf, HIDP, sw2t, HIDP, eo, Dn, Tn, Dn, HIDn);

  // 11) final residual sum -> d_out
  k_final<<<2048, 256, 0, stream>>>(hbuf, moe, eo, dout, (size_t)Tn * Dn);
}
